// Feature_embedding_pre_process_pos_88751204205177
// MI455X (gfx1250) — matrix operand order repaired and checked by reading, not compiled, not run on hardware
//
#include <hip/hip_runtime.h>
#include <hip/hip_bf16.h>
#include <math.h>

typedef __attribute__((ext_vector_type(16))) _Float16 v16h;
typedef __attribute__((ext_vector_type(8)))  float    v8f;
typedef __attribute__((ext_vector_type(4)))  float    v4f;

// Problem constants (from the reference)
#define N_DOCS      128
#define FEATURE_DIM 220
#define K_PAD       224   // FEATURE_DIM padded to multiple of 32
#define HIDDEN      768

// ---------------------------------------------------------------------------
// Kernel 1: W (HIDDEN x FEATURE_DIM, f32, row-major) -> f16, K padded to 224
// with zeros so the GEMM B-fragment loads need no masking.
// ---------------------------------------------------------------------------
__global__ void convert_w_kernel(const float* __restrict__ W,
                                 _Float16* __restrict__ Wh) {
    int idx = blockIdx.x * blockDim.x + threadIdx.x;
    if (idx >= HIDDEN * K_PAD) return;
    int row = idx / K_PAD;        // output-neuron index n
    int col = idx - row * K_PAD;  // k
    float v = (col < FEATURE_DIM) ? W[(size_t)row * FEATURE_DIM + col] : 0.0f;
    Wh[idx] = (_Float16)v;
}

// ---------------------------------------------------------------------------
// Kernel 2: gather rows + signed-log transform.
// One wave (32 lanes) per row; lanes stride over the 220 features (coalesced).
// Writes f32 feature_batch (output #2 region of d_out).
// ---------------------------------------------------------------------------
__global__ void gather_transform_kernel(const int* __restrict__ ids,
                                        const float* __restrict__ table,
                                        float* __restrict__ fb,
                                        int n_rows) {
    int row  = blockIdx.x * 8 + (threadIdx.x >> 5);
    int lane = threadIdx.x & 31;
    if (row >= n_rows) return;
    int q = ids[2 * row + 0];
    int d = ids[2 * row + 1];
    const float* src = table + ((size_t)q * N_DOCS + d) * FEATURE_DIM;
    float*       dst = fb + (size_t)row * FEATURE_DIM;
    for (int c = lane; c < FEATURE_DIM; c += 32) {
        float x = src[c];
        dst[c] = copysignf(log1pf(fabsf(x)), x);   // sign(x)*log(|x|+1)
    }
}

// ---------------------------------------------------------------------------
// A-fragment builders (16x32 f16 A tile, same k order as the B fragment):
//   lane l: row m0+(l&15); elements 0..15 = K[k0 + 16h + j] (contiguous)
//   (h = l>>4). Loads f32 feature_batch, converts to f16 on the fly.
// ---------------------------------------------------------------------------
__device__ __forceinline__ v16h load_a_frag(const float* __restrict__ Arow,
                                            int k0, int h) {
    const v4f* p = (const v4f*)(Arow + k0 + h * 16);
    const v4f* q = (const v4f*)(Arow + k0 + 8 + h * 16);
    v4f x0 = p[0], x1 = p[1];
    v4f y0 = q[0], y1 = q[1];
    v16h a;
#pragma unroll
    for (int j = 0; j < 4; ++j) {
        a[j]      = (_Float16)x0[j];
        a[4 + j]  = (_Float16)x1[j];
        a[8 + j]  = (_Float16)y0[j];
        a[12 + j] = (_Float16)y1[j];
    }
    return a;
}

__device__ __forceinline__ v16h load_a_frag_tail(const float* __restrict__ Arow,
                                                 int k0, int h) {
    v16h a;
    int c0 = k0 + h * 16;
    int c1 = k0 + 8 + h * 16;
#pragma unroll
    for (int j = 0; j < 8; ++j) {
        int ca = c0 + j, cb = c1 + j;
        a[j]     = (_Float16)((ca < FEATURE_DIM) ? Arow[ca] : 0.0f);
        a[8 + j] = (_Float16)((cb < FEATURE_DIM) ? Arow[cb] : 0.0f);
    }
    return a;
}

// B-fragment (32x16 f16 B tile): lane l holds column n0+(l&15); 16 contiguous
// K-halves starting at k0 + 16*(l>>4). Contiguous 32-byte aligned load from
// the zero-padded f16 W.
__device__ __forceinline__ v16h load_b_frag(const _Float16* __restrict__ Wcol,
                                            int k0) {
    return *(const v16h*)(Wcol + k0);
}

// ---------------------------------------------------------------------------
// Kernel 3: WMMA GEMM  out = relu(A @ W^T + b)
//   A: (M=65536, K=220) f32 (feature_batch), W: padded f16 (768 x 224)
//   Block = 256 threads = 8 waves. Each wave: 16(M) x 64(N) tile.
//   Grid: (M/16/8 = 512, HIDDEN/64 = 12).
// ---------------------------------------------------------------------------
__global__ void wmma_gemm_kernel(const float* __restrict__ A,
                                 const _Float16* __restrict__ Wh,
                                 const float* __restrict__ bias,
                                 float* __restrict__ out) {
    const int lane = threadIdx.x & 31;
    const int wave = threadIdx.x >> 5;
    const int h    = lane >> 4;   // 0 or 1
    const int l16  = lane & 15;

    const int m0 = (blockIdx.x * 8 + wave) * 16;
    const int n0 = blockIdx.y * 64;

    const float* Arow = A + (size_t)(m0 + l16) * FEATURE_DIM;
    // Base of this lane's B column data for the 4 N-subtiles (stride 16 cols).
    const _Float16* Wb0 = Wh + (size_t)(n0 + 0 * 16 + l16) * K_PAD + h * 16;
    const _Float16* Wb1 = Wh + (size_t)(n0 + 1 * 16 + l16) * K_PAD + h * 16;
    const _Float16* Wb2 = Wh + (size_t)(n0 + 2 * 16 + l16) * K_PAD + h * 16;
    const _Float16* Wb3 = Wh + (size_t)(n0 + 3 * 16 + l16) * K_PAD + h * 16;

    v8f acc0 = {}, acc1 = {}, acc2 = {}, acc3 = {};

#pragma unroll
    for (int k0 = 0; k0 < 192; k0 += 32) {
        v16h a  = load_a_frag(Arow, k0, h);
        v16h b0 = load_b_frag(Wb0, k0);
        v16h b1 = load_b_frag(Wb1, k0);
        v16h b2 = load_b_frag(Wb2, k0);
        v16h b3 = load_b_frag(Wb3, k0);
        acc0 = __builtin_amdgcn_wmma_f32_16x16x32_f16(false, a, false, b0, (short)0, acc0, false, false);
        acc1 = __builtin_amdgcn_wmma_f32_16x16x32_f16(false, a, false, b1, (short)0, acc1, false, false);
        acc2 = __builtin_amdgcn_wmma_f32_16x16x32_f16(false, a, false, b2, (short)0, acc2, false, false);
        acc3 = __builtin_amdgcn_wmma_f32_16x16x32_f16(false, a, false, b3, (short)0, acc3, false, false);
    }
    {   // K tail: 192..219 (A masked; B already zero-padded)
        const int k0 = 192;
        v16h a  = load_a_frag_tail(Arow, k0, h);
        v16h b0 = load_b_frag(Wb0, k0);
        v16h b1 = load_b_frag(Wb1, k0);
        v16h b2 = load_b_frag(Wb2, k0);
        v16h b3 = load_b_frag(Wb3, k0);
        acc0 = __builtin_amdgcn_wmma_f32_16x16x32_f16(false, a, false, b0, (short)0, acc0, false, false);
        acc1 = __builtin_amdgcn_wmma_f32_16x16x32_f16(false, a, false, b1, (short)0, acc1, false, false);
        acc2 = __builtin_amdgcn_wmma_f32_16x16x32_f16(false, a, false, b2, (short)0, acc2, false, false);
        acc3 = __builtin_amdgcn_wmma_f32_16x16x32_f16(false, a, false, b3, (short)0, acc3, false, false);
    }

    // Bias (per-N, one scalar per lane per subtile)
    const float bv0 = bias[n0 + 0 * 16 + l16];
    const float bv1 = bias[n0 + 1 * 16 + l16];
    const float bv2 = bias[n0 + 2 * 16 + l16];
    const float bv3 = bias[n0 + 3 * 16 + l16];

    // D layout: element r of acc <-> (M = m0 + r + 8h, N = n0 + nn*16 + l16)
#pragma unroll
    for (int r = 0; r < 8; ++r) {
        const int m = m0 + r + h * 8;
        float* orow = out + (size_t)m * HIDDEN + n0 + l16;
        orow[0]  = fmaxf(acc0[r] + bv0, 0.0f);
        orow[16] = fmaxf(acc1[r] + bv1, 0.0f);
        orow[32] = fmaxf(acc2[r] + bv2, 0.0f);
        orow[48] = fmaxf(acc3[r] + bv3, 0.0f);
    }
}

// ---------------------------------------------------------------------------
// Launch
// ---------------------------------------------------------------------------
extern "C" void kernel_launch(void* const* d_in, const int* in_sizes, int n_in,
                              void* d_out, int out_size, void* d_ws, size_t ws_size,
                              hipStream_t stream) {
    const int*   ids   = (const int*)d_in[0];     // (N_IDS, 2) int32
    const float* table = (const float*)d_in[1];   // (2000, 128, 220) f32
    const float* W     = (const float*)d_in[2];   // (768, 220) f32
    const float* bias  = (const float*)d_in[3];   // (768,) f32

    const int n_rows = in_sizes[0] / 2;           // 65536

    float* out_embeds = (float*)d_out;                         // (n_rows, 768)
    float* fb         = out_embeds + (size_t)n_rows * HIDDEN;  // (n_rows, 220)

    _Float16* Wh = (_Float16*)d_ws;               // (768, 224) f16, 344 KB

    // 1) W -> padded f16
    convert_w_kernel<<<(HIDDEN * K_PAD + 255) / 256, 256, 0, stream>>>(W, Wh);

    // 2) gather + signed-log transform (writes output #2)
    gather_transform_kernel<<<(n_rows + 7) / 8, 256, 0, stream>>>(ids, table, fb, n_rows);

    // 3) WMMA GEMM + bias + relu (writes output #1)
    dim3 grid(n_rows / (16 * 8), HIDDEN / 64, 1);
    wmma_gemm_kernel<<<grid, 256, 0, stream>>>(fb, Wh, bias, out_embeds);
}
